// Attention_Model_2259152797767
// MI455X (gfx1250) — matrix operand order repaired and checked by reading, not compiled, not run on hardware
//
#include <hip/hip_runtime.h>
#include <hip/hip_bf16.h>

typedef _Float16 half_t;
typedef __attribute__((ext_vector_type(16))) _Float16 v16h;
typedef __attribute__((ext_vector_type(8)))  _Float16 v8h;
typedef __attribute__((ext_vector_type(8)))  float    v8f;
typedef __attribute__((ext_vector_type(4)))  float    v4f;
typedef __attribute__((ext_vector_type(4)))  unsigned int v4u;

#define BATCH 32
#define SEQ   512
#define EMB   512
#define NHEAD 8
#define HDIM  64
#define MTOT  (BATCH * SEQ)   // 16384

// all LDS row strides are multiples of 8 halfs (16 B) so fragments load as b128
#define ASTR  48   // A panels  [64][48]
#define BSTR  48   // B panels  [n][k] = [64][48]
#define KSTR  72   // K panel   [32 keys][72]   (144 B rows, 16B-aligned)
#define VTSTR 48   // V^T panel [64 dh][48]
#define PSTR  48   // P slabs   [8][16][48]
#define AJSTR 48   // adj panel [128][48] bytes

#define WMMA_F32_F16(a, b, c) \
  __builtin_amdgcn_wmma_f32_16x16x32_f16(false, (a), false, (b), (short)0, (c), false, false)

__device__ __forceinline__ int lane_id() { return (int)(threadIdx.x & 31u); }

__device__ __forceinline__ v16h cat8(v8h lo, v8h hi) {
  return __builtin_shufflevector(lo, hi, 0, 1, 2, 3, 4, 5, 6, 7,
                                         8, 9, 10, 11, 12, 13, 14, 15);
}

// A-fragment 16x32, src row-major [row][k], lda multiple of 8 halfs.
// Per lane this is two contiguous 16-byte runs -> 2x b128 loads.
__device__ __forceinline__ v16h frag_a(const half_t* __restrict__ src, int lda) {
  const int lane  = lane_id();
  const int row   = lane & 15;
  const int khalf = (lane >> 4) << 4;            // 0 or 16
  const half_t* p = src + row * lda + khalf;
  return cat8(*(const v8h*)p, *(const v8h*)(p + 8));
}

// B-fragment 32x16 read from B^T storage, src row-major [n][k], ld mult of 8.
// Per lane one contiguous 32-byte run -> 2x b128 loads.
__device__ __forceinline__ v16h frag_bT(const half_t* __restrict__ src, int ld) {
  const int lane  = lane_id();
  const int col   = lane & 15;
  const int kbase = (lane >> 4) << 4;            // 0 or 16
  const half_t* p = src + col * ld + kbase;
  return cat8(*(const v8h*)p, *(const v8h*)(p + 8));
}

// ---------------------------------------------------------------------------
// Kernel 1: fused QKV projection.  out = x @ W^T + b -> f16 [b][h][n][dh]
// ---------------------------------------------------------------------------
__global__ __launch_bounds__(256) void qkv_proj_kernel(
    const float* __restrict__ x,
    const float* __restrict__ Wq, const float* __restrict__ Wk, const float* __restrict__ Wv,
    const float* __restrict__ bq, const float* __restrict__ bk, const float* __restrict__ bv,
    half_t* __restrict__ Q, half_t* __restrict__ K, half_t* __restrict__ V) {
  __shared__ half_t As[64][ASTR];
  __shared__ half_t Bs[64][BSTR];   // W rows verbatim: Bs[n][k]

  const float* W; const float* bias; half_t* dst;
  if (blockIdx.z == 0)      { W = Wq; bias = bq; dst = Q; }
  else if (blockIdx.z == 1) { W = Wk; bias = bk; dst = K; }
  else                      { W = Wv; bias = bv; dst = V; }

  const int m0 = blockIdx.x * 64;
  const int n0 = blockIdx.y * 64;
  const int tid = threadIdx.x;
  const int w = tid >> 5;
  const int lane = tid & 31;
  const int col16 = lane & 15;
  const int rbase = (lane >> 4) << 3;
  const int mt  = w & 3;
  const int ntb = (w >> 2) * 32;

  // staging coords: each thread owns one 8-float run (32 B read, 16 B LDS write)
  const int sr = tid >> 2;            // 0..63
  const int sc = (tid & 3) * 8;       // 0,8,16,24

  v8f c0, c1;
#pragma unroll
  for (int i = 0; i < 8; ++i) { c0[i] = 0.f; c1[i] = 0.f; }

  for (int k0 = 0; k0 < EMB; k0 += 32) {
    __syncthreads();
    {
      const float* gp = x + (size_t)(m0 + sr) * EMB + k0 + sc;
      v4f f0 = *(const v4f*)gp, f1 = *(const v4f*)(gp + 4);
      v8h h8;
#pragma unroll
      for (int i = 0; i < 4; ++i) { h8[i] = (half_t)f0[i]; h8[4 + i] = (half_t)f1[i]; }
      *(v8h*)&As[sr][sc] = h8;
    }
    {
      const float* gp = W + (size_t)(n0 + sr) * EMB + k0 + sc;
      v4f f0 = *(const v4f*)gp, f1 = *(const v4f*)(gp + 4);
      v8h h8;
#pragma unroll
      for (int i = 0; i < 4; ++i) { h8[i] = (half_t)f0[i]; h8[4 + i] = (half_t)f1[i]; }
      *(v8h*)&Bs[sr][sc] = h8;
    }
    __syncthreads();

    v16h a  = frag_a(&As[mt * 16][0], ASTR);
    v16h b0 = frag_bT(&Bs[ntb][0], BSTR);
    v16h b1 = frag_bT(&Bs[ntb + 16][0], BSTR);
    c0 = WMMA_F32_F16(a, b0, c0);
    c1 = WMMA_F32_F16(a, b1, c1);
  }

#pragma unroll
  for (int t = 0; t < 2; ++t) {
    const v8f c = t ? c1 : c0;
    const int e = n0 + ntb + t * 16 + col16;
    const float bb = bias[e];
    const int hh = e >> 6, dh = e & 63;
#pragma unroll
    for (int r = 0; r < 8; ++r) {
      const int m = m0 + mt * 16 + rbase + r;
      const int bi = m >> 9, sq = m & 511;
      dst[(((size_t)bi * NHEAD + hh) * SEQ + sq) * HDIM + dh] = (half_t)(c[r] + bb);
    }
  }
}

// ---------------------------------------------------------------------------
// Kernel 2: fused masked flash attention. 8 waves/block, wave owns 16 queries.
// K panel staged via CDNA5 async global->LDS copy (ASYNCcnt); V staged
// transposed so P.V B-fragments are contiguous b128 loads.
// ---------------------------------------------------------------------------
__global__ __launch_bounds__(256) void attn_kernel(
    const unsigned char* __restrict__ adj,
    const half_t* __restrict__ Qw, const half_t* __restrict__ Kw,
    const half_t* __restrict__ Vw, half_t* __restrict__ Ow) {
  __shared__ half_t Ks[32][KSTR];            // [key][dh]  (B^T layout for Q.K^T)
  __shared__ half_t VsT[64][VTSTR];          // [dh][key]  (B^T layout for P.V)
  __shared__ half_t Ps[8][16][PSTR];         // per-wave P slab
  __shared__ unsigned char Adjs[128][AJSTR]; // adj panel, key-contiguous

  const int tid = threadIdx.x;
  const int w = tid >> 5;
  const int lane = tid & 31;
  const int col16 = lane & 15;
  const int rbase = (lane >> 4) << 3;

  const int b = blockIdx.z;
  const int h = blockIdx.y;
  const int qblk = blockIdx.x * 128;
  const int q0 = qblk + w * 16;
  const size_t bh = (size_t)b * NHEAD + h;

  // Q fragments (DH=64 -> two K=32 chunks) straight from global as b128s.
  const half_t* Qp = Qw + (bh * SEQ + q0) * HDIM;
  const v16h qf0 = frag_a(Qp, HDIM);
  const v16h qf1 = frag_a(Qp + 32, HDIM);

  v8f o[4];
#pragma unroll
  for (int t = 0; t < 4; ++t)
#pragma unroll
    for (int i = 0; i < 8; ++i) o[t][i] = 0.f;

  float mr[8], lr[8];
#pragma unroll
  for (int r = 0; r < 8; ++r) { mr[r] = -1e30f; lr[r] = 0.f; }

  const float scale = 0.125f;  // 1/sqrt(64)

  // staging coords
  const int skey = tid >> 3;            // 0..31
  const int sdh  = (tid & 7) * 8;       // 0,8,...,56
  const int aq   = tid >> 1;            // 0..127
  const int aks  = (tid & 1) * 16;      // 0 or 16

  for (int kb = 0; kb < SEQ; kb += 32) {
    __syncthreads();  // previous panel fully consumed
    const half_t* Kg = Kw + (bh * SEQ + kb) * HDIM;
    const half_t* Vg = Vw + (bh * SEQ + kb) * HDIM;

    // adj panel: 16 contiguous bytes per thread
    *(v4u*)&Adjs[aq][aks] =
        *(const v4u*)(adj + (size_t)(qblk + aq) * SEQ + kb + aks);

    // K panel: one async 16-byte global->LDS copy per thread (ASYNCcnt path)
    {
      unsigned lds_off = (unsigned)(uintptr_t)&Ks[skey][sdh];
      const half_t* gk = Kg + skey * HDIM + sdh;
      asm volatile("global_load_async_to_lds_b128 %0, %1, off"
                   :: "v"(lds_off), "v"(gk) : "memory");
    }
    // V panel: read 16 B, store transposed
    {
      v8h vv = *(const v8h*)(Vg + skey * HDIM + sdh);
#pragma unroll
      for (int i = 0; i < 8; ++i) VsT[sdh + i][skey] = vv[i];
    }
    if (kb + 32 < SEQ) {
      __builtin_prefetch(Kg + 32 * HDIM + tid * 8, 0, 1);
      __builtin_prefetch(Vg + 32 * HDIM + tid * 8, 0, 1);
    }
    asm volatile("s_wait_asynccnt 0x0" ::: "memory");
    __syncthreads();

    // S = (Q K^T) * scale for two 16-key tiles; mask via LDS adj panel.
    float p[2][8];
#pragma unroll
    for (int t2 = 0; t2 < 2; ++t2) {
      v8f s;
#pragma unroll
      for (int i = 0; i < 8; ++i) s[i] = 0.f;
      v16h bk0 = frag_bT(&Ks[t2 * 16][0], KSTR);    // dh 0..31
      v16h bk1 = frag_bT(&Ks[t2 * 16][32], KSTR);   // dh 32..63
      s = WMMA_F32_F16(qf0, bk0, s);
      s = WMMA_F32_F16(qf1, bk1, s);
      const int klocal = t2 * 16 + col16;
#pragma unroll
      for (int r = 0; r < 8; ++r) {
        const int qlocal = w * 16 + rbase + r;
        const float sv = s[r] * scale;
        p[t2][r] = Adjs[qlocal][klocal] ? sv : -1e30f;
      }
    }

    // Online softmax; row stats align with C-tile layout (row = rbase + r).
#pragma unroll
    for (int r = 0; r < 8; ++r) {
      float mx = fmaxf(p[0][r], p[1][r]);
#pragma unroll
      for (int d = 1; d < 16; d <<= 1) mx = fmaxf(mx, __shfl_xor(mx, d, 32));
      const float mnew = fmaxf(mr[r], mx);
      const float e0 = __expf(p[0][r] - mnew);
      const float e1 = __expf(p[1][r] - mnew);
      p[0][r] = e0; p[1][r] = e1;
      float rs = e0 + e1;
#pragma unroll
      for (int d = 1; d < 16; d <<= 1) rs += __shfl_xor(rs, d, 32);
      const float fac = __expf(mr[r] - mnew);  // ->0 once a real key appears
      lr[r] = lr[r] * fac + rs;
      mr[r] = mnew;
#pragma unroll
      for (int t = 0; t < 4; ++t) o[t][r] *= fac;
    }

    // Re-stage P (C layout -> A layout) through this wave's LDS slab.
#pragma unroll
    for (int t2 = 0; t2 < 2; ++t2)
#pragma unroll
      for (int r = 0; r < 8; ++r)
        Ps[w][rbase + r][t2 * 16 + col16] = (half_t)p[t2][r];

    v16h ap = frag_a(&Ps[w][0][0], PSTR);
#pragma unroll
    for (int t = 0; t < 4; ++t) {
      v16h bv = frag_bT(&VsT[t * 16][0], VTSTR);  // contiguous b128 loads
      o[t] = WMMA_F32_F16(ap, bv, o[t]);
    }
  }

  // Normalize and store O f16 in [b][n][h*64+dh] for the output GEMM.
#pragma unroll
  for (int t = 0; t < 4; ++t) {
    const int e = h * HDIM + t * 16 + col16;
#pragma unroll
    for (int r = 0; r < 8; ++r) {
      const int sq = q0 + rbase + r;
      Ow[((size_t)b * SEQ + sq) * EMB + e] = (half_t)(o[t][r] / lr[r]);
    }
  }
}

// ---------------------------------------------------------------------------
// Kernel 3: output projection.  out = O @ Wo^T + bo, f32 result.
// ---------------------------------------------------------------------------
__global__ __launch_bounds__(256) void out_proj_kernel(
    const half_t* __restrict__ Ocat, const float* __restrict__ Wo,
    const float* __restrict__ bo, float* __restrict__ out) {
  __shared__ half_t As[64][ASTR];
  __shared__ half_t Bs[64][BSTR];

  const int m0 = blockIdx.x * 64;
  const int n0 = blockIdx.y * 64;
  const int tid = threadIdx.x;
  const int w = tid >> 5;
  const int lane = tid & 31;
  const int col16 = lane & 15;
  const int rbase = (lane >> 4) << 3;
  const int mt  = w & 3;
  const int ntb = (w >> 2) * 32;

  const int sr = tid >> 2;
  const int sc = (tid & 3) * 8;

  v8f c0, c1;
#pragma unroll
  for (int i = 0; i < 8; ++i) { c0[i] = 0.f; c1[i] = 0.f; }

  for (int k0 = 0; k0 < EMB; k0 += 32) {
    __syncthreads();
    // A is already f16: one b128 copy per thread
    *(v8h*)&As[sr][sc] = *(const v8h*)(Ocat + (size_t)(m0 + sr) * EMB + k0 + sc);
    {
      const float* gp = Wo + (size_t)(n0 + sr) * EMB + k0 + sc;
      v4f f0 = *(const v4f*)gp, f1 = *(const v4f*)(gp + 4);
      v8h h8;
#pragma unroll
      for (int i = 0; i < 4; ++i) { h8[i] = (half_t)f0[i]; h8[4 + i] = (half_t)f1[i]; }
      *(v8h*)&Bs[sr][sc] = h8;
    }
    __syncthreads();

    v16h a  = frag_a(&As[mt * 16][0], ASTR);
    v16h b0 = frag_bT(&Bs[ntb][0], BSTR);
    v16h b1 = frag_bT(&Bs[ntb + 16][0], BSTR);
    c0 = WMMA_F32_F16(a, b0, c0);
    c1 = WMMA_F32_F16(a, b1, c1);
  }

#pragma unroll
  for (int t = 0; t < 2; ++t) {
    const v8f c = t ? c1 : c0;
    const int e = n0 + ntb + t * 16 + col16;
    const float bb = bo[e];
#pragma unroll
    for (int r = 0; r < 8; ++r) {
      const int m = m0 + mt * 16 + rbase + r;
      out[(size_t)m * EMB + e] = c[r] + bb;
    }
  }
}

// ---------------------------------------------------------------------------
extern "C" void kernel_launch(void* const* d_in, const int* in_sizes, int n_in,
                              void* d_out, int out_size, void* d_ws, size_t ws_size,
                              hipStream_t stream) {
  (void)in_sizes; (void)n_in; (void)out_size; (void)ws_size;
  const float*         x   = (const float*)d_in[0];
  const unsigned char* adj = (const unsigned char*)d_in[1];  // numpy bool = 1 byte
  const float* Wq = (const float*)d_in[2];
  const float* Wk = (const float*)d_in[3];
  const float* Wv = (const float*)d_in[4];
  const float* bq = (const float*)d_in[5];
  const float* bk = (const float*)d_in[6];
  const float* bv = (const float*)d_in[7];
  const float* Wo = (const float*)d_in[8];
  const float* bo = (const float*)d_in[9];

  const size_t QKV_ELEMS = (size_t)BATCH * NHEAD * SEQ * HDIM;  // 8M halfs each
  half_t* Qws = (half_t*)d_ws;
  half_t* Kws = Qws + QKV_ELEMS;
  half_t* Vws = Kws + QKV_ELEMS;
  half_t* Ows = Vws + QKV_ELEMS;   // 64 MB total workspace

  dim3 g1(MTOT / 64, EMB / 64, 3);
  qkv_proj_kernel<<<g1, 256, 0, stream>>>(x, Wq, Wk, Wv, bq, bk, bv, Qws, Kws, Vws);

  dim3 g2(SEQ / 128, NHEAD, BATCH);
  attn_kernel<<<g2, 256, 0, stream>>>(adj, Qws, Kws, Vws, Ows);

  dim3 g3(MTOT / 64, EMB / 64, 1);
  out_proj_kernel<<<g3, 256, 0, stream>>>(Ows, Wo, bo, (float*)d_out);
}
